// UpdateNode_85744727097816
// MI455X (gfx1250) — matrix operand order repaired and checked by reading, not compiled, not run on hardware
//
// CDNA5 (gfx1250) equivariant GNN message-passing layer.
// All dense matmuls via v_wmma_f32_16x16x32_bf16 (M=16 edges/nodes per wave32
// tile, f32 accumulate). Weights pre-packed into WMMA B-fragment layout and
// staged into LDS via the Tensor Data Mover (TENSOR_LOAD_TO_LDS + TENSORcnt);
// intermediates kept in bf16 to halve bandwidth.
#include <hip/hip_runtime.h>
#include <hip/hip_bf16.h>

typedef __attribute__((ext_vector_type(16))) __bf16 v16bf;
typedef __attribute__((ext_vector_type(8)))  __bf16 v8bf;
typedef __attribute__((ext_vector_type(8)))  float  v8f;
typedef __attribute__((ext_vector_type(4)))  unsigned int u32x4;
typedef __attribute__((ext_vector_type(8)))  int i32x8;
typedef __attribute__((ext_vector_type(4)))  int i32x4;

#define NND 10000
#define NED 160000

__device__ __constant__ float RSQ3 = 0.5773502691896258f;
__device__ __constant__ float RSQ2 = 0.7071067811865475f;

// Fragment bases (in units of fragments; 1 fragment = 32 lanes * 16 bf16 = 1KB)
enum {
  FB_SS0 = 0,    // 8x12
  FB_VV0 = 96,   // 4x12
  FB_SV1 = 144,  // 8x4
  FB_VS1 = 176,  // 4x4
  FB_VV1 = 192,  // 4x4
  FB_PS  = 208,  // 4x8
  FB_ENV = 240,  // 2x12
  FB_PV  = 264,  // 2x4
  FB_RS  = 272,  // 4x8
  FB_RV  = 304,  // 2x4
  FB_TOTAL = 312
};

#if defined(__HIP_DEVICE_COMPILE__) && defined(__has_builtin)
#if __has_builtin(__builtin_amdgcn_tensor_load_to_lds) && \
    __has_builtin(__builtin_amdgcn_s_wait_tensorcnt)
#define USE_TDM 1
#endif
#endif

// Stage `nbytes` (multiple of 8, < 512KB) of weight fragments from global to
// LDS offset 0. TDM path: one wave issues a 1-D D# (data_size=8B,
// tile_dim0=tensor_dim0=stride0=nbytes/8), waits TENSORcnt, block barriers.
__device__ __forceinline__ void stage_weights(char* smem, const void* gsrc,
                                              unsigned nbytes) {
#ifdef USE_TDM
  if (threadIdx.x < 32) {
    unsigned n8 = nbytes >> 3;  // elements of 8 bytes
    unsigned long long ga = (unsigned long long)gsrc;
    u32x4 g0;
    g0[0] = 1u;                                   // count=1, user mode
    g0[1] = 0u;                                   // lds_addr = 0 (dynamic LDS base)
    g0[2] = (unsigned)(ga & 0xFFFFFFFFu);         // global_addr[31:0]
    g0[3] = (unsigned)((ga >> 32) & 0x01FFFFFFu)  // global_addr[56:32]
            | (2u << 30);                         // type = 2 ("image")
    i32x8 g1;
    g1[0] = 0x00030000;                           // data_size = 3 (8B)
    g1[1] = (int)((n8 & 0xFFFFu) << 16);          // tensor_dim0[15:0]
    g1[2] = (int)((n8 >> 16) & 0xFFFFu)           // tensor_dim0[31:16]
            | (1 << 16);                          // tensor_dim1 = 1
    g1[3] = (int)((n8 & 0xFFFFu) << 16);          // tile_dim0 = n8
    g1[4] = 0;                                    // tile_dim1/2 unused (1-D)
    g1[5] = (int)n8;                              // tensor_dim0_stride[31:0]
    g1[6] = (int)((n8 & 0xFFFFu) << 16);          // stride0[47:32]=0 | stride1 lo
    g1[7] = (int)(n8 >> 16);                      // stride1 hi
    i32x4 z4 = {0, 0, 0, 0};
#if __clang_major__ >= 23
    i32x8 z8 = {0, 0, 0, 0, 0, 0, 0, 0};
    __builtin_amdgcn_tensor_load_to_lds(g0, g1, z4, z4, z8, 0);
#else
    __builtin_amdgcn_tensor_load_to_lds(g0, g1, z4, z4, 0);
#endif
    __builtin_amdgcn_s_wait_tensorcnt(0);
  }
  __syncthreads();
#else
  const uint4* srcp = (const uint4*)gsrc;
  uint4* dstp = (uint4*)smem;
  for (unsigned i = threadIdx.x; i < nbytes / 16; i += blockDim.x)
    dstp[i] = srcp[i];
  __syncthreads();
#endif
}

__device__ __forceinline__ v8f wmma_bf(v16bf a, v16bf b, v8f c) {
  return __builtin_amdgcn_wmma_f32_16x16x32_bf16(false, a, false, b, (short)0, c,
                                                 false, false);
}

// B fragment: lane holds column n = lane&15, elements i<8 -> K = 8*(lane>>4)+i, i>=8 -> K = 16+8*(lane>>4)+(i-8)
__device__ __forceinline__ v16bf ld_frag(const __bf16* base, int f, int lane) {
  return *(const v16bf*)(base + (size_t)f * 512 + (size_t)lane * 16);
}

// A fragment from bf16 row-major storage; rowbase already includes kt*32.
// elements 0..7 -> K = half*8 + i ; elements 8..15 -> K = half*8 + 16 + i
__device__ __forceinline__ v16bf ld_a_bf(const __bf16* rowbase, int half) {
  v8bf lo = *(const v8bf*)(rowbase + half * 8);
  v8bf hi = *(const v8bf*)(rowbase + half * 8 + 16);
  v16bf r;
#pragma unroll
  for (int i = 0; i < 8; ++i) { r[i] = lo[i]; r[8 + i] = hi[i]; }
  return r;
}

__device__ __forceinline__ v16bf ld_a_f32(const float* rowbase, int half) {
  v16bf r;
#pragma unroll
  for (int i = 0; i < 8; ++i) {
    r[i]     = (__bf16)rowbase[half * 8 + i];
    r[8 + i] = (__bf16)rowbase[half * 8 + 16 + i];
  }
  return r;
}

__device__ __forceinline__ v16bf fscale(v16bf a, float s) {
  v16bf r;
#pragma unroll
  for (int i = 0; i < 16; ++i) r[i] = (__bf16)((float)a[i] * s);
  return r;
}

// ---------------------------------------------------------------- weight pack
__global__ void pack_b_kernel(const float* __restrict__ W, __bf16* __restrict__ dst,
                              int nkt, int nnt, int Nn) {
  int t = blockIdx.x * blockDim.x + threadIdx.x;
  int total = nkt * nnt * 32;
  if (t >= total) return;
  int lane = t & 31;
  int f = t >> 5;
  int kt = f / nnt, nt = f - kt * nnt;
  int n = nt * 16 + (lane & 15);
  int kb = kt * 32 + (lane >> 4) * 8;
  v16bf v;
#pragma unroll
  for (int i = 0; i < 16; ++i) v[i] = (__bf16)W[(size_t)(kb + i + (i >= 8 ? 8 : 0)) * Nn + n];
  *(v16bf*)(dst + (size_t)f * 512 + (size_t)lane * 16) = v;
}

// ------------------------------------------------------------------- sep LN
__global__ void sep_ln_kernel(const float* __restrict__ feat, int nrows,
                              const float* __restrict__ g_s,
                              const float* __restrict__ b_s,
                              const float* __restrict__ g_v,
                              __bf16* __restrict__ out_s,
                              __bf16* __restrict__ out_v) {
  int w = (blockIdx.x * blockDim.x + threadIdx.x) >> 5;
  int lane = threadIdx.x & 31;
  if (w >= nrows) return;
  const float* row = feat + (size_t)w * 320;
  float x[4];
  float s = 0.f, ss = 0.f;
#pragma unroll
  for (int i = 0; i < 4; ++i) {
    x[i] = row[lane + i * 32];
    s += x[i];
    ss += x[i] * x[i];
  }
#pragma unroll
  for (int m = 1; m < 32; m <<= 1) { s += __shfl_xor(s, m); ss += __shfl_xor(ss, m); }
  float mu = s * (1.f / 128.f);
  float var = ss * (1.f / 128.f) - mu * mu;
  float rinv = rsqrtf(var + 1e-5f);
#pragma unroll
  for (int i = 0; i < 4; ++i) {
    int c = lane + i * 32;
    out_s[(size_t)w * 128 + c] = (__bf16)((x[i] - mu) * rinv * g_s[c] + b_s[c]);
  }
  float v[2][3];
  float vs = 0.f;
#pragma unroll
  for (int j = 0; j < 2; ++j) {
    int c = lane + j * 32;
#pragma unroll
    for (int d = 0; d < 3; ++d) {
      v[j][d] = row[128 + c * 3 + d];
      vs += v[j][d] * v[j][d];
    }
  }
#pragma unroll
  for (int m = 1; m < 32; m <<= 1) vs += __shfl_xor(vs, m);
  float rv = rsqrtf(vs * (1.f / 64.f) + 1e-5f);
#pragma unroll
  for (int j = 0; j < 2; ++j) {
    int c = lane + j * 32;
#pragma unroll
    for (int d = 0; d < 3; ++d)
      out_v[(size_t)d * nrows * 64 + (size_t)w * 64 + c] =
          (__bf16)(v[j][d] * rv * g_v[c]);
  }
}

// -------------------------------------------------------- residual (nodes)
__global__ void residual_kernel(const float* __restrict__ nf,
                                const __bf16* __restrict__ wfrag,
                                float* __restrict__ out) {
  int w = (blockIdx.x * blockDim.x + threadIdx.x) >> 5;
  int lane = threadIdx.x & 31;
  if (w >= NND / 16) return;
  int n0 = w * 16, half = lane >> 4, mrow = lane & 15;
  const float* arow = nf + (size_t)(n0 + mrow) * 320;
  v8f acc[8] = {};
#pragma unroll
  for (int kt = 0; kt < 4; ++kt) {
    v16bf a = ld_a_f32(arow + kt * 32, half);
#pragma unroll
    for (int nt = 0; nt < 8; ++nt)
      acc[nt] = wmma_bf(a, ld_frag(wfrag, FB_RS + kt * 8 + nt, lane), acc[nt]);
  }
#pragma unroll
  for (int nt = 0; nt < 8; ++nt)
#pragma unroll
    for (int r = 0; r < 8; ++r)
      out[(size_t)(n0 + r + 8 * half) * 320 + nt * 16 + mrow] = acc[nt][r];
#pragma unroll
  for (int d = 0; d < 3; ++d) {
    v8f av[4] = {};
#pragma unroll
    for (int kt = 0; kt < 2; ++kt) {
      v16bf a;
#pragma unroll
      for (int i = 0; i < 8; ++i) {
        int k1 = kt * 32 + half * 8 + i;
        int k2 = k1 + 16;
        a[i]     = (__bf16)arow[128 + k1 * 3 + d];
        a[8 + i] = (__bf16)arow[128 + k2 * 3 + d];
      }
#pragma unroll
      for (int nt = 0; nt < 4; ++nt)
        av[nt] = wmma_bf(a, ld_frag(wfrag, FB_RV + kt * 4 + nt, lane), av[nt]);
    }
#pragma unroll
    for (int nt = 0; nt < 4; ++nt)
#pragma unroll
      for (int r = 0; r < 8; ++r)
        out[(size_t)(n0 + r + 8 * half) * 320 + 128 + 3 * (nt * 16 + mrow) + d] =
            av[nt][r];
  }
}

// -------------------------------------------- per-edge message (out0/out_v)
__global__ void __launch_bounds__(256, 1) edge_msg_kernel(
    const float* __restrict__ esh, const int* __restrict__ ec,
    const __bf16* __restrict__ wfrag_g, const __bf16* __restrict__ ns,
    const __bf16* __restrict__ nv, const __bf16* __restrict__ es,
    const __bf16* __restrict__ ev, __bf16* __restrict__ sbuf,
    __bf16* __restrict__ vbuf) {
  extern __shared__ __align__(32) char smem[];
  __bf16* wl = (__bf16*)smem;
  stage_weights(smem, wfrag_g, 208 * 1024);
  int lane = threadIdx.x & 31, half = lane >> 4, mrow = lane & 15;
  int wid = (blockIdx.x * blockDim.x + threadIdx.x) >> 5;
  int nw = (gridDim.x * blockDim.x) >> 5;
  for (int tile = wid; tile < NED / 16; tile += nw) {
    int e0 = tile * 16;
    int e = e0 + mrow;
    int src = ec[e];
    float ash0 = esh[e * 4 + 0];
    float asx = esh[e * 4 + 1], asy = esh[e * 4 + 2], asz = esh[e * 4 + 3];
    v16bf aS[8];
#pragma unroll
    for (int kt = 0; kt < 8; ++kt)
      aS[kt] = (kt < 4) ? ld_a_bf(ns + (size_t)src * 128 + kt * 32, half)
                        : ld_a_bf(es + (size_t)e * 128 + (kt - 4) * 32, half);
    v16bf aV[3][4];
#pragma unroll
    for (int d = 0; d < 3; ++d)
#pragma unroll
      for (int kt = 0; kt < 4; ++kt)
        aV[d][kt] =
            (kt < 2)
                ? ld_a_bf(nv + (size_t)d * NND * 64 + (size_t)src * 64 + kt * 32, half)
                : ld_a_bf(ev + (size_t)d * NED * 64 + (size_t)e * 64 + (kt - 2) * 32,
                          half);
    v8f o[12] = {};
    v8f T[4] = {};
#pragma unroll
    for (int kt = 0; kt < 8; ++kt) {
      v16bf as = fscale(aS[kt], ash0);  // (S * sh0) for W_ss0 path
#pragma unroll
      for (int nt = 0; nt < 12; ++nt)
        o[nt] = wmma_bf(as, ld_frag(wl, FB_SS0 + kt * 12 + nt, lane), o[nt]);
#pragma unroll
      for (int nt = 0; nt < 4; ++nt)  // raw S for W_sv1 path
        T[nt] = wmma_bf(aS[kt], ld_frag(wl, FB_SV1 + kt * 4 + nt, lane), T[nt]);
    }
#pragma unroll
    for (int kt = 0; kt < 4; ++kt) {  // dotV = (V . sh1)/sqrt(3)
      v16bf ad;
#pragma unroll
      for (int i = 0; i < 16; ++i)
        ad[i] = (__bf16)(((float)aV[0][kt][i] * asx + (float)aV[1][kt][i] * asy +
                          (float)aV[2][kt][i] * asz) * RSQ3);
#pragma unroll
      for (int nt = 0; nt < 12; ++nt)
        o[nt] = wmma_bf(ad, ld_frag(wl, FB_VV0 + kt * 12 + nt, lane), o[nt]);
    }
    // s = silu(out0[:, :128]) -> bf16 buffer
#pragma unroll
    for (int nt = 0; nt < 8; ++nt)
#pragma unroll
      for (int r = 0; r < 8; ++r) {
        float x = o[nt][r];
        sbuf[(size_t)(e0 + r + 8 * half) * 128 + nt * 16 + mrow] =
            (__bf16)(x / (1.f + __expf(-x)));
      }
    v8f gate[4];
#pragma unroll
    for (int g = 0; g < 4; ++g)
#pragma unroll
      for (int r = 0; r < 8; ++r) gate[g][r] = 1.f / (1.f + __expf(-o[8 + g][r]));
    float dsh[4][8];  // per-D-row edge_sh values (row m = r + 8*half)
#pragma unroll
    for (int r = 0; r < 8; ++r) {
      int ee = e0 + r + 8 * half;
#pragma unroll
      for (int q = 0; q < 4; ++q) dsh[q][r] = esh[ee * 4 + q];
    }
#pragma unroll
    for (int d = 0; d < 3; ++d) {
      v8f a4[4] = {};
#pragma unroll
      for (int kt = 0; kt < 4; ++kt)  // V @ W_vs1
#pragma unroll
        for (int nt = 0; nt < 4; ++nt)
          a4[nt] = wmma_bf(aV[d][kt], ld_frag(wl, FB_VS1 + kt * 4 + nt, lane), a4[nt]);
#pragma unroll
      for (int nt = 0; nt < 4; ++nt)  // * sh0 (row scalar, D side)
#pragma unroll
        for (int r = 0; r < 8; ++r) a4[nt][r] *= dsh[0][r];
      const int d1 = (d + 1) % 3, d2 = (d + 2) % 3;
      const float sA = (d2 == 0) ? asx : (d2 == 1) ? asy : asz;
      const float sB = (d1 == 0) ? asx : (d1 == 1) ? asy : asz;
#pragma unroll
      for (int kt = 0; kt < 4; ++kt) {  // cross(V, sh1)_d / sqrt(2) @ W_vv1
        v16bf ax;
#pragma unroll
        for (int i = 0; i < 16; ++i)
          ax[i] = (__bf16)(((float)aV[d1][kt][i] * sA - (float)aV[d2][kt][i] * sB) *
                           RSQ2);
#pragma unroll
        for (int nt = 0; nt < 4; ++nt)
          a4[nt] = wmma_bf(ax, ld_frag(wl, FB_VV1 + kt * 4 + nt, lane), a4[nt]);
      }
#pragma unroll
      for (int nt = 0; nt < 4; ++nt)
#pragma unroll
        for (int r = 0; r < 8; ++r) {
          float val = (a4[nt][r] + T[nt][r] * dsh[1 + d][r]) * gate[nt][r];
          vbuf[(size_t)d * NED * 64 + (size_t)(e0 + r + 8 * half) * 64 + nt * 16 +
               mrow] = (__bf16)val;
        }
    }
  }
}

// ------------------------------- project, envelope, scatter-add aggregation
__global__ void __launch_bounds__(256, 1) edge_agg_kernel(
    const float* __restrict__ lat, const int* __restrict__ ec,
    const __bf16* __restrict__ wfrag_g, const __bf16* __restrict__ sbuf,
    const __bf16* __restrict__ vbuf, float* __restrict__ out) {
  extern __shared__ __align__(32) char smem[];
  __bf16* wl = (__bf16*)smem;  // frags [FB_PS, FB_PS+64)
  stage_weights(smem, wfrag_g + (size_t)FB_PS * 512, 64 * 1024);
  const int L_PS = 0, L_ENV = 32, L_PV = 56;
  int lane = threadIdx.x & 31, half = lane >> 4, mrow = lane & 15;
  int wid = (blockIdx.x * blockDim.x + threadIdx.x) >> 5;
  int nw = (gridDim.x * blockDim.x) >> 5;
  for (int tile = wid; tile < NED / 16; tile += nw) {
    int e0 = tile * 16;
    int e = e0 + mrow;
    int dnode[8];
#pragma unroll
    for (int r = 0; r < 8; ++r) dnode[r] = ec[e0 + r + 8 * half];
    v8f w[12] = {};
#pragma unroll
    for (int kt = 0; kt < 2; ++kt) {  // w = latents @ W_env
      v16bf a = ld_a_f32(lat + (size_t)e * 64 + kt * 32, half);
#pragma unroll
      for (int nt = 0; nt < 12; ++nt)
        w[nt] = wmma_bf(a, ld_frag(wl, L_ENV + kt * 12 + nt, lane), w[nt]);
    }
    v8f s2[8] = {};
#pragma unroll
    for (int kt = 0; kt < 4; ++kt) {  // s @ Wp_s
      v16bf a = ld_a_bf(sbuf + (size_t)e * 128 + kt * 32, half);
#pragma unroll
      for (int nt = 0; nt < 8; ++nt)
        s2[nt] = wmma_bf(a, ld_frag(wl, L_PS + kt * 8 + nt, lane), s2[nt]);
    }
#pragma unroll
    for (int nt = 0; nt < 8; ++nt)
#pragma unroll
      for (int r = 0; r < 8; ++r)
        atomicAdd(out + (size_t)dnode[r] * 320 + nt * 16 + mrow,
                  s2[nt][r] * w[nt][r] * 0.25f);
#pragma unroll
    for (int d = 0; d < 3; ++d) {
      v8f v4[4] = {};
#pragma unroll
      for (int kt = 0; kt < 2; ++kt) {  // v @ Wp_v
        v16bf a =
            ld_a_bf(vbuf + (size_t)d * NED * 64 + (size_t)e * 64 + kt * 32, half);
#pragma unroll
        for (int nt = 0; nt < 4; ++nt)
          v4[nt] = wmma_bf(a, ld_frag(wl, L_PV + kt * 4 + nt, lane), v4[nt]);
      }
#pragma unroll
      for (int nt = 0; nt < 4; ++nt)
#pragma unroll
        for (int r = 0; r < 8; ++r)
          atomicAdd(out + (size_t)dnode[r] * 320 + 128 + 3 * (nt * 16 + mrow) + d,
                    v4[nt][r] * w[8 + nt][r] * 0.25f);
    }
  }
}

// ---------------------------------------------------------------- launcher
extern "C" void kernel_launch(void* const* d_in, const int* in_sizes, int n_in,
                              void* d_out, int out_size, void* d_ws, size_t ws_size,
                              hipStream_t stream) {
  (void)in_sizes; (void)n_in; (void)out_size; (void)ws_size;
  const float* latents       = (const float*)d_in[0];
  const float* node_features = (const float*)d_in[1];
  const float* edge_features = (const float*)d_in[2];
  const float* edge_sh       = (const float*)d_in[3];
  const int*   edge_index    = (const int*)d_in[4];  // row 0 = source nodes
  const float* g_s_n = (const float*)d_in[7];
  const float* b_s_n = (const float*)d_in[8];
  const float* g_v_n = (const float*)d_in[9];
  const float* g_s_e = (const float*)d_in[10];
  const float* b_s_e = (const float*)d_in[11];
  const float* g_v_e = (const float*)d_in[12];
  const float* W_ss0 = (const float*)d_in[13];
  const float* W_vv0 = (const float*)d_in[14];
  const float* W_sv1 = (const float*)d_in[15];
  const float* W_vs1 = (const float*)d_in[16];
  const float* W_vv1 = (const float*)d_in[17];
  const float* Wp_s  = (const float*)d_in[18];
  const float* Wp_v  = (const float*)d_in[19];
  const float* W_env = (const float*)d_in[20];
  const float* Wr_s  = (const float*)d_in[21];
  const float* Wr_v  = (const float*)d_in[22];
  float* out = (float*)d_out;

  char* ws = (char*)d_ws;
  constexpr size_t WS_FRAG = 0;                                   // 319488 B
  constexpr size_t WS_NS = 327680;                                // N*128 bf16
  constexpr size_t WS_NV = WS_NS + (size_t)NND * 128 * 2;         // 3*N*64 bf16
  constexpr size_t WS_ES = WS_NV + (size_t)3 * NND * 64 * 2;      // E*128 bf16
  constexpr size_t WS_EV = WS_ES + (size_t)NED * 128 * 2;         // 3*E*64 bf16
  constexpr size_t WS_SB = WS_EV + (size_t)3 * NED * 64 * 2;      // E*128 bf16
  constexpr size_t WS_VB = WS_SB + (size_t)NED * 128 * 2;         // 3*E*64 bf16
  __bf16* wfrag = (__bf16*)(ws + WS_FRAG);
  __bf16* ns    = (__bf16*)(ws + WS_NS);
  __bf16* nv    = (__bf16*)(ws + WS_NV);
  __bf16* es    = (__bf16*)(ws + WS_ES);
  __bf16* ev    = (__bf16*)(ws + WS_EV);
  __bf16* sbuf  = (__bf16*)(ws + WS_SB);
  __bf16* vbuf  = (__bf16*)(ws + WS_VB);

  auto pack = [&](const float* W, int fb, int nkt, int nnt, int Nn) {
    int total = nkt * nnt * 32;
    pack_b_kernel<<<(total + 255) / 256, 256, 0, stream>>>(
        W, wfrag + (size_t)fb * 512, nkt, nnt, Nn);
  };
  pack(W_ss0, FB_SS0, 8, 12, 192);
  pack(W_vv0, FB_VV0, 4, 12, 192);
  pack(W_sv1, FB_SV1, 8, 4, 64);
  pack(W_vs1, FB_VS1, 4, 4, 64);
  pack(W_vv1, FB_VV1, 4, 4, 64);
  pack(Wp_s,  FB_PS,  4, 8, 128);
  pack(W_env, FB_ENV, 2, 12, 192);
  pack(Wp_v,  FB_PV,  2, 4, 64);
  pack(Wr_s,  FB_RS,  4, 8, 128);
  pack(Wr_v,  FB_RV,  2, 4, 64);

  sep_ln_kernel<<<(NND + 7) / 8, 256, 0, stream>>>(node_features, NND, g_s_n, b_s_n,
                                                   g_v_n, ns, nv);
  sep_ln_kernel<<<(NED + 7) / 8, 256, 0, stream>>>(edge_features, NED, g_s_e, b_s_e,
                                                   g_v_e, es, ev);
  residual_kernel<<<(NND / 16 * 32 + 255) / 256, 256, 0, stream>>>(node_features,
                                                                   wfrag, out);
  edge_msg_kernel<<<256, 256, 208 * 1024, stream>>>(edge_sh, edge_index, wfrag, ns,
                                                    nv, es, ev, sbuf, vbuf);
  edge_agg_kernel<<<512, 256, 64 * 1024, stream>>>(latents, edge_index, wfrag, sbuf,
                                                   vbuf, out);
}
